// UpsampleConv_82695300317580
// MI455X (gfx1250) — matrix operand order repaired and checked by reading, not compiled, not run on hardware
//
#include <hip/hip_runtime.h>
#include <stdint.h>

typedef __attribute__((ext_vector_type(16))) _Float16 v16h;
typedef __attribute__((ext_vector_type(8)))  float    v8f;
typedef __attribute__((ext_vector_type(4)))  uint32_t u32x4;
typedef __attribute__((ext_vector_type(8)))  uint32_t u32x8;

#define B_    8
#define C_    128
#define HOUT  128
#define WOUT  128
#define O_    128
#define NEXP  8
#define KDIM  1152              // 9*C  (kk = j*128 + c, j = ky*3+kx)
#define NCELL 32768             // B*64*64 half-res routing cells
#define NPIX  131072            // B*128*128 output pixels
#define PIXCAP 132096           // NPIX + NEXP*128 (segments 128-aligned)
#define NTILEWG 1032            // PIXCAP/128 workgroups
#define PACKN  1179648          // NEXP*KDIM*O_ packed f16 weight elements

// workspace layout (bytes)
#define WS_COUNTS   0           // u32[8]
#define WS_CURSOR   32          // u32[8]
#define WS_PADOFF   64          // u32[9]
#define WS_EXPERT   256         // u8[32768]
#define WS_PIXLIST  33280       // u32[132096]
#define WS_PACKW    561664      // _Float16[1179648]  (~2.92MB total ws use)

// ---------------- init: pixel list sentinel + counters ----------------
__global__ void k0_init(uint32_t* pixList, uint32_t* counts, uint32_t* cursor) {
  int t = blockIdx.x * blockDim.x + threadIdx.x;
  if (t < PIXCAP) pixList[t] = 0xFFFFFFFFu;
  if (t < NEXP) { counts[t] = 0u; cursor[t] = 0u; }
}

// ---------------- pack weights into WMMA B-fragment order -------------
// packW[((k*36+kstep)*8+nt)*512 + lane*16 + h] = W[o=nt*16+(lane&15)][c][k][ky][kx]
// with kk = kstep*32 + (h&8)*2 + (lane>>4)*8 + (h&7) = (ky*3+kx)*128 + c
__global__ void k_pack(const float* __restrict__ w, _Float16* __restrict__ pw) {
  int e = blockIdx.x * 256 + threadIdx.x;      // 4608*256 == PACKN exactly
  int h    = e & 15;
  int lane = (e >> 4) & 31;
  int nt   = (e >> 9) & 7;
  int ke   = e >> 12;                           // k*36 + kstep
  int k    = ke / 36;
  int kstep= ke - k * 36;
  int kk   = kstep * 32 + ((h & 8) << 1) + ((lane >> 4) << 3) + (h & 7);
  int j    = kk >> 7;
  int cch  = kk & 127;
  int ky   = (j * 43) >> 7;                     // j/3 for j<9
  int kx   = j - ky * 3;
  int o    = nt * 16 + (lane & 15);
  float v = w[(((o * C_ + cch) * NEXP + k) * 3 + ky) * 3 + kx];
  pw[e] = (_Float16)v;
}

// ---------------- routing: argmax of coupler logits per half-res cell -
__global__ void k_route(const float* __restrict__ x, const float* __restrict__ cw,
                        const float* __restrict__ cb,
                        uint8_t* __restrict__ expertIdx, uint32_t* __restrict__ counts) {
  int t = blockIdx.x * blockDim.x + threadIdx.x;   // cell in [0, NCELL)
  int b = t >> 12, rem = t & 4095;                 // 64*64 = 4096
  const float* xp = x + ((size_t)(b * C_) << 12) + rem;
  float acc[NEXP];
#pragma unroll
  for (int k = 0; k < NEXP; ++k) acc[k] = cb[k];
  for (int c = 0; c < C_; ++c) {
    float xv = xp[(size_t)c << 12];
#pragma unroll
    for (int k = 0; k < NEXP; ++k) acc[k] += xv * cw[k * C_ + c];
  }
  int best = 0;
#pragma unroll
  for (int k = 1; k < NEXP; ++k) if (acc[k] > acc[best]) best = k;  // first-max tiebreak
  expertIdx[t] = (uint8_t)best;
  atomicAdd(&counts[best], 4u);                    // 4 full-res pixels per cell
}

// ---------------- prefix offsets (128-aligned segments) ---------------
__global__ void k_offsets(const uint32_t* counts, uint32_t* padOff) {
  if (blockIdx.x == 0 && threadIdx.x == 0) {
    uint32_t o = 0;
    for (int k = 0; k < NEXP; ++k) { padOff[k] = o; o += (counts[k] + 127u) & ~127u; }
    padOff[NEXP] = o;
  }
}

// ---------------- scatter pixels into per-expert bins -----------------
__global__ void k_scatter(const uint8_t* __restrict__ expertIdx,
                          const uint32_t* __restrict__ padOff,
                          uint32_t* cursor, uint32_t* pixList) {
  int p = blockIdx.x * blockDim.x + threadIdx.x;   // [0, NPIX)
  int b = p >> 14, y = (p >> 7) & 127, xx = p & 127;
  int cell = (b << 12) + ((y >> 1) << 6) + (xx >> 1);
  int k = expertIdx[cell];
  uint32_t pos = atomicAdd(&cursor[k], 1u);
  pixList[padOff[k] + pos] = (uint32_t)p;
}

// ---------------- grouped im2col GEMM via WMMA + TDM ------------------
union HFrag { v16h v; uint4 q[2]; };

__global__ __launch_bounds__(256) void k_gemm(
    const float* __restrict__ xin, const _Float16* __restrict__ packW,
    const float* __restrict__ bias, const uint32_t* __restrict__ padOff,
    const uint32_t* __restrict__ pixList, float* __restrict__ out) {

  __shared__ __align__(16) _Float16 ldsA[128 * 72];  // 128 rows x 64 halves, pad to 72
  __shared__ __align__(16) _Float16 ldsB[8192];      // 2 ksteps x 8 nt x 32 lanes x 16 halves
  __shared__ uint32_t ldsPix[128];

  uint32_t base = blockIdx.x * 128u;
  if (base >= padOff[8]) return;                     // uniform across block
  int k = 0;
  while (base >= padOff[k + 1]) ++k;

  int t = threadIdx.x;
  if (t < 128) ldsPix[t] = pixList[base + t];
  __syncthreads();

  int lane = t & 31, wave = t >> 5;
  int laneHi = lane >> 4, nlo = lane & 15;
  int rowG = t & 127;                                // gather row (coalesced-ish)
  int idx0 = t >> 7;                                 // 0 or 1

  size_t expertBase = (size_t)k * (36 * 8 * 512);    // halves

  // TDM descriptor invariants (2D degenerate = flat 16KB tile of 8B elems)
  uint32_t ldsBAddr = (uint32_t)(uintptr_t)&ldsB[0]; // LDS byte offset (low 32 bits)
  u32x4 g0;
  u32x8 g1;
  g1[0] = 3u << 16;                                  // data_size = 3 (8 bytes)
  g1[1] = (2048u & 0xFFFFu) << 16;                   // tensor_dim0[15:0]
  g1[2] = (2048u >> 16) | (1u << 16);                // tensor_dim0[31:16] | tensor_dim1[15:0]=1
  g1[3] = 0u | (2048u << 16);                        // tensor_dim1[31:16] | tile_dim0=2048
  g1[4] = 1u;                                        // tile_dim1=1, tile_dim2=0
  g1[5] = 2048u;                                     // tensor_dim0_stride[31:0]
  g1[6] = 0u;                                        // stride0[47:32] | stride1[15:0]
  g1[7] = 0u;                                        // stride1[47:16]

  v8f acc[8];
#pragma unroll
  for (int nt = 0; nt < 8; ++nt)
    acc[nt] = v8f{0.f, 0.f, 0.f, 0.f, 0.f, 0.f, 0.f, 0.f};

  for (int ch = 0; ch < 18; ++ch) {
    __syncthreads();                                 // prior-chunk ds reads done (wait+signal)

    // --- stage B tile: Tensor Data Mover, one DMA per workgroup ---
    if (wave == 0) {
      uint64_t gaddr = (uint64_t)(uintptr_t)(packW + expertBase + (size_t)ch * 8192);
      g0[0] = 1u;                                    // count=1, user descriptor
      g0[1] = ldsBAddr;
      g0[2] = (uint32_t)gaddr;
      g0[3] = (uint32_t)(gaddr >> 32) | 0x80000000u; // type=2 ("image")
      asm volatile("tensor_load_to_lds %0, %1" :: "s"(g0), "s"(g1) : "memory");
    }

    // --- stage A tile: fused 2x nearest-upsample + zero-pad im2col gather ---
    uint32_t pG = ldsPix[rowG];
    int pb = (int)(pG >> 14), py = (int)((pG >> 7) & 127), px = (int)(pG & 127);
#pragma unroll 4
    for (int i = 0; i < 32; ++i) {
      int idx = idx0 + i * 2;                        // 0..63 within chunk
      int kkg = ch * 64 + idx;
      int j = kkg >> 7, cch = kkg & 127;
      int ky = (j * 43) >> 7, kx = j - ky * 3;
      _Float16 hv = (_Float16)0.f;
      if (pG != 0xFFFFFFFFu) {
        int uy = py + ky - 1, ux = px + kx - 1;
        if ((unsigned)uy < 128u && (unsigned)ux < 128u) {
          float f = xin[(((pb << 7) + cch) << 12) + ((uy >> 1) << 6) + (ux >> 1)];
          hv = (_Float16)f;
        }
      }
      ldsA[rowG * 72 + idx] = hv;
    }

    __builtin_amdgcn_s_wait_tensorcnt(0);            // TDM done (no-op for waves 1..7)
    __syncthreads();

    // --- compute: per kstep load ALL 8 B frags (forces distinct regs),
    //     then 8 back-to-back WMMAs behind a single dscnt wait ---
#pragma unroll
    for (int s = 0; s < 2; ++s) {
      HFrag a;
      const uint4* pa = (const uint4*)&ldsA[(wave * 16 + nlo) * 72 + s * 32 + laneHi * 8];
      a.q[0] = pa[0];                                // K {0..7} / {8..15}
      a.q[1] = pa[2];                                // K {16..23} / {24..31}

      HFrag bf[8];
#pragma unroll
      for (int nt = 0; nt < 8; ++nt) {
        const uint4* pn = (const uint4*)&ldsB[((s * 8 + nt) * 32 + lane) * 16];
        bf[nt].q[0] = pn[0];
        bf[nt].q[1] = pn[1];
      }
#pragma unroll
      for (int nt = 0; nt < 8; ++nt) {
        acc[nt] = __builtin_amdgcn_wmma_f32_16x16x32_f16(
            false, a.v, false, bf[nt].v, (short)0, acc[nt], false, false);
      }
    }
  }

  // --- scatter-store D (C layout: vgpr r -> M=r / M=8+r; lane -> N) + bias ---
  uint32_t offr[8]; bool valid[8];
#pragma unroll
  for (int r = 0; r < 8; ++r) {
    uint32_t p = ldsPix[wave * 16 + r + laneHi * 8];
    valid[r] = (p != 0xFFFFFFFFu);
    uint32_t b2 = p >> 14, y2 = (p >> 7) & 127, x2 = p & 127;
    offr[r] = b2 * 2097152u + y2 * 128u + x2;        // b*O*H*W + y*W + x
  }
#pragma unroll
  for (int nt = 0; nt < 8; ++nt) {
    int n = nt * 16 + nlo;
    float bv = bias[n];
#pragma unroll
    for (int r = 0; r < 8; ++r) {
      if (valid[r]) out[offr[r] + (uint32_t)n * 16384u] = acc[nt][r] + bv;
    }
  }
}

// ---------------- launch --------------------------------------------
extern "C" void kernel_launch(void* const* d_in, const int* in_sizes, int n_in,
                              void* d_out, int out_size, void* d_ws, size_t ws_size,
                              hipStream_t stream) {
  const float* x     = (const float*)d_in[0];   // [8,128,64,64]
  const float* convW = (const float*)d_in[1];   // [128,128,8,3,3]
  const float* convB = (const float*)d_in[2];   // [128]
  const float* coupW = (const float*)d_in[3];   // [8,128,1,1]
  const float* coupB = (const float*)d_in[4];   // [8]
  float* out = (float*)d_out;

  char* ws = (char*)d_ws;
  uint32_t* counts   = (uint32_t*)(ws + WS_COUNTS);
  uint32_t* cursor   = (uint32_t*)(ws + WS_CURSOR);
  uint32_t* padOff   = (uint32_t*)(ws + WS_PADOFF);
  uint8_t*  expertIx = (uint8_t*)(ws + WS_EXPERT);
  uint32_t* pixList  = (uint32_t*)(ws + WS_PIXLIST);
  _Float16* packW    = (_Float16*)(ws + WS_PACKW);

  k0_init  <<<(PIXCAP + 255) / 256, 256, 0, stream>>>(pixList, counts, cursor);
  k_pack   <<<PACKN / 256,          256, 0, stream>>>(convW, packW);
  k_route  <<<NCELL / 256,          256, 0, stream>>>(x, coupW, coupB, expertIx, counts);
  k_offsets<<<1,                      1, 0, stream>>>(counts, padOff);
  k_scatter<<<NPIX / 256,           256, 0, stream>>>(expertIx, padOff, cursor, pixList);
  k_gemm   <<<NTILEWG,              256, 0, stream>>>(x, packW, convB, padOff, pixList, out);
}
